// ImplicitQueryAttention_26242250178739
// MI455X (gfx1250) — matrix operand order repaired and checked by reading, not compiled, not run on hardware
//
#include <hip/hip_runtime.h>
#include <hip/hip_bf16.h>

typedef __attribute__((ext_vector_type(16))) _Float16 v16h;
typedef __attribute__((ext_vector_type(8)))  float    v8f;
typedef __attribute__((ext_vector_type(4)))  unsigned int v4u;

#define BB 4
#define NN 2048
#define DD 1024

// LDS leading dimensions (units: elements), padded & 16B-aligned rows
#define LDK 1032   // halves, K-panel 16 x 1024
#define LDS_S 2052 // floats, score panel 16 x 2048
#define LDP 2056   // halves, P panel 16 x 2048

__device__ __forceinline__ v8f wmma16x16x32(v16h a, v16h b, v8f c) {
  // D = A(16x32 f16) * B(32x16 f16) + C(f32)
  return __builtin_amdgcn_wmma_f32_16x16x32_f16(
      /*neg_a=*/false, a, /*neg_b=*/false, b,
      /*c_mod=*/(short)0, c, /*reuse_a=*/false, /*reuse_b=*/false);
}

// A fragment (16x32, f16) from row-major src: row M = lane&15,
// halves 0..7 = K kb..kb+7, halves 8..15 = K 16+kb..16+kb+7, kb = 8*(lane>=16).
__device__ __forceinline__ v16h load_fragA(const _Float16* base, int stride,
                                           int row0, int k0) {
  int lane = threadIdx.x & 31;
  int r = lane & 15;
  int kb = (lane >> 4) << 3;
  const _Float16* p = base + (size_t)(row0 + r) * stride + k0 + kb;
  v16h out;
  ((v4u*)&out)[0] = *(const v4u*)p;
  ((v4u*)&out)[1] = *(const v4u*)(p + 16);
  return out;
}

// B fragment (32x16, f16) where `base` holds B^T row-major (rows = columns of B):
// lane column N = lane&15, halves 0..7 = K kb..kb+7, halves 8..15 = K 16+kb..16+kb+7, kb = 8*(lane>=16).
__device__ __forceinline__ v16h load_fragB(const _Float16* base, int stride,
                                           int col0, int k0) {
  int lane = threadIdx.x & 31;
  int c = lane & 15;
  int ko = (lane >> 4) << 3;
  const _Float16* p = base + (size_t)(col0 + c) * stride + k0 + ko;
  v16h out;
  ((v4u*)&out)[0] = *(const v4u*)p;
  ((v4u*)&out)[1] = *(const v4u*)(p + 16);
  return out;
}

// ---------------- prep kernels ----------------

__global__ void cvt_f32_f16(const float* __restrict__ in,
                            _Float16* __restrict__ out, int n) {
  int i = blockIdx.x * 256 + threadIdx.x;
  if (i < n) out[i] = (_Float16)in[i];
}

// xT[b][d][n] = (f16) x[b][n][d]
__global__ void transpose_to_f16(const float* __restrict__ x,
                                 _Float16* __restrict__ xT, int N, int D) {
  __shared__ float tile[32][33];
  int b = blockIdx.z;
  int d0 = blockIdx.x * 32, n0 = blockIdx.y * 32;
  const float* xb = x + (size_t)b * N * D;
  for (int i = threadIdx.y; i < 32; i += 8)
    tile[i][threadIdx.x] = xb[(size_t)(n0 + i) * D + d0 + threadIdx.x];
  __syncthreads();
  _Float16* xTb = xT + (size_t)b * N * D;
  for (int i = threadIdx.y; i < 32; i += 8)
    xTb[(size_t)(d0 + i) * N + n0 + threadIdx.x] = (_Float16)tile[threadIdx.x][i];
}

// ---------------- WMMA GEMM: C[b][n][e] = sum_d A[b][n][d] * W[e][d] ----------------
// one wave per block; wave computes a 16x64 output tile
__global__ __launch_bounds__(32) void gemm_abT_f16(
    const _Float16* __restrict__ A, const _Float16* __restrict__ W,
    _Float16* __restrict__ C, int N, int D, int E) {
  int b = blockIdx.z;
  int n0 = blockIdx.y * 16;
  int e0 = blockIdx.x * 64;
  const _Float16* Ab = A + (size_t)b * N * D;
  v8f acc0 = {}, acc1 = {}, acc2 = {}, acc3 = {};
  for (int k0 = 0; k0 < D; k0 += 32) {
    v16h a = load_fragA(Ab, D, n0, k0);
    acc0 = wmma16x16x32(a, load_fragB(W, D, e0,      k0), acc0);
    acc1 = wmma16x16x32(a, load_fragB(W, D, e0 + 16, k0), acc1);
    acc2 = wmma16x16x32(a, load_fragB(W, D, e0 + 32, k0), acc2);
    acc3 = wmma16x16x32(a, load_fragB(W, D, e0 + 48, k0), acc3);
  }
  int lane = threadIdx.x & 31;
  int col = lane & 15, ro = (lane >> 4) * 8;
  _Float16* Cb = C + (size_t)b * N * E;
  v8f* accs[4] = {&acc0, &acc1, &acc2, &acc3};
#pragma unroll
  for (int j = 0; j < 4; ++j)
#pragma unroll
    for (int r = 0; r < 8; ++r)
      Cb[(size_t)(n0 + ro + r) * E + e0 + j * 16 + col] = (_Float16)((*accs[j])[r]);
}

// ---------------- fused attention iteration ----------------
// One block = 16 query(K)-rows. 8 waves, 226KB dynamic LDS.
// S = Kblk(16x1024) @ Q^T -> LDS f32; softmax (unnormalized P f16, 1/sum folded
// into output); y = P(16x2048) @ x(2048x1024) using xT rows as B^T.
__global__ __launch_bounds__(256) void attn_iter(
    const _Float16* __restrict__ Kh,   // [B][N][D]
    const _Float16* __restrict__ Qh,   // [B][N][D]
    const _Float16* __restrict__ xTh,  // [B][D][N]
    float* __restrict__ y32,           // [B][N][D]  (d_out)
    _Float16* __restrict__ yh)         // [B][N][D]  (next iterate, f16)
{
  extern __shared__ char smem_raw[];
  _Float16* Kbl = (_Float16*)smem_raw;                         // 16*LDK halves
  float*    S   = (float*)(smem_raw + 16 * LDK * 2);           // 16*LDS_S floats
  _Float16* P   = (_Float16*)((char*)S + 16 * LDS_S * 4);      // 16*LDP halves
  float*    red = (float*)((char*)P + 16 * LDP * 2);           // 16*16 floats
  float*    rowinv = red + 256;                                // 16 floats

  int tileIdx = blockIdx.x;
  int b  = tileIdx / (NN / 16);
  int n0 = (tileIdx % (NN / 16)) * 16;
  int tid = threadIdx.x;
  int wave = tid >> 5;
  int lane = tid & 31;

  // stage K-row panel into LDS (16 rows x 1024 f16), 16B vectors
  {
    const _Float16* src = Kh + ((size_t)b * NN + n0) * DD;
    for (int i = tid; i < 16 * (DD / 8); i += 256) {
      int r = i / (DD / 8), c8 = i % (DD / 8);
      *(v4u*)(Kbl + r * LDK + c8 * 8) = *(const v4u*)(src + (size_t)r * DD + c8 * 8);
    }
  }
  __syncthreads();

  // ---- Phase A: score panel S[16][2048] ----
  {
    const _Float16* Qb = Qh + (size_t)b * NN * DD;
    v8f acc[16] = {};
    for (int k0 = 0; k0 < DD; k0 += 32) {
      v16h a = load_fragA(Kbl, LDK, 0, k0);
#pragma unroll
      for (int t = 0; t < 16; ++t) {
        int m0 = (wave * 16 + t) * 16;
        acc[t] = wmma16x16x32(a, load_fragB(Qb, DD, m0, k0), acc[t]);
      }
    }
    int col = lane & 15, ro = (lane >> 4) * 8;
    for (int t = 0; t < 16; ++t) {
      int m0 = (wave * 16 + t) * 16;
#pragma unroll
      for (int r = 0; r < 8; ++r)
        S[(ro + r) * LDS_S + m0 + col] = acc[t][r];
    }
  }
  __syncthreads();

  // ---- Phase B: softmax stats; store unnormalized exp() as f16 P ----
  {
    int r = tid >> 4, c0 = tid & 15;
    float mx = -3.4e38f;
    for (int c = c0; c < NN; c += 16) mx = fmaxf(mx, S[r * LDS_S + c]);
    red[r * 16 + c0] = mx;
    __syncthreads();
    mx = red[r * 16];
    for (int i = 1; i < 16; ++i) mx = fmaxf(mx, red[r * 16 + i]);
    __syncthreads();
    float sum = 0.f;
    for (int c = c0; c < NN; c += 16) {
      float p = __expf(S[r * LDS_S + c] - mx);
      sum += p;
      P[r * LDP + c] = (_Float16)p;
    }
    red[r * 16 + c0] = sum;
    __syncthreads();
    if (c0 == 0) {
      float s = 0.f;
      for (int i = 0; i < 16; ++i) s += red[r * 16 + i];
      rowinv[r] = 1.0f / s;
    }
  }
  __syncthreads();

  // ---- Phase C: y[16][1024] = P @ x, via xT rows; fold 1/sum at writeback ----
  {
    const _Float16* xTb = xTh + (size_t)b * (size_t)DD * NN;
    v8f acc[8] = {};
    for (int m0 = 0; m0 < NN; m0 += 32) {
      v16h a = load_fragA(P, LDP, 0, m0);
#pragma unroll
      for (int j = 0; j < 8; ++j) {
        int d0 = (wave * 8 + j) * 16;
        acc[j] = wmma16x16x32(a, load_fragB(xTb, NN, d0, m0), acc[j]);
      }
    }
    int col = lane & 15, ro = (lane >> 4) * 8;
    float*    yb  = y32 + ((size_t)b * NN + n0) * DD;
    _Float16* yhb = yh  + ((size_t)b * NN + n0) * DD;
    for (int j = 0; j < 8; ++j) {
      int d0 = (wave * 8 + j) * 16;
#pragma unroll
      for (int r = 0; r < 8; ++r) {
        float v = acc[j][r] * rowinv[ro + r];
        yb [(size_t)(ro + r) * DD + d0 + col] = v;
        yhb[(size_t)(ro + r) * DD + d0 + col] = (_Float16)v;
      }
    }
  }
}

// ---------------- host launcher ----------------

extern "C" void kernel_launch(void* const* d_in, const int* in_sizes, int n_in,
                              void* d_out, int out_size, void* d_ws, size_t ws_size,
                              hipStream_t stream) {
  const float* x  = (const float*)d_in[0];
  const float* WQ = (const float*)d_in[1];
  const float* WK = (const float*)d_in[2];
  float* y32 = (float*)d_out;

  const size_t nxd = (size_t)BB * NN * DD;       // 8M elements
  char* ws = (char*)d_ws;
  _Float16* x_h  = (_Float16*)ws; ws += nxd * 2;
  _Float16* xT_h = (_Float16*)ws; ws += nxd * 2;
  _Float16* WQ_h = (_Float16*)ws; ws += (size_t)DD * DD * 2;
  _Float16* WK_h = (_Float16*)ws; ws += (size_t)DD * DD * 2;
  _Float16* K_h  = (_Float16*)ws; ws += nxd * 2;
  _Float16* Q_h  = (_Float16*)ws; ws += nxd * 2;
  _Float16* y_h  = (_Float16*)ws; ws += nxd * 2;
  (void)ws_size; (void)in_sizes; (void)n_in; (void)out_size;

  // one-time prep
  cvt_f32_f16<<<(unsigned)((nxd + 255) / 256), 256, 0, stream>>>(x, x_h, (int)nxd);
  cvt_f32_f16<<<(DD * DD + 255) / 256, 256, 0, stream>>>(WQ, WQ_h, DD * DD);
  cvt_f32_f16<<<(DD * DD + 255) / 256, 256, 0, stream>>>(WK, WK_h, DD * DD);
  transpose_to_f16<<<dim3(DD / 32, NN / 32, BB), dim3(32, 8), 0, stream>>>(x, xT_h, NN, DD);

  // K = x @ W_K^T  (once)
  gemm_abT_f16<<<dim3(DD / 64, NN / 16, BB), 32, 0, stream>>>(x_h, WK_h, K_h, NN, DD, DD);

  const size_t smem = (size_t)16 * LDK * 2 + (size_t)16 * LDS_S * 4 +
                      (size_t)16 * LDP * 2 + (256 + 16) * 4;  // ~226 KB

  const _Float16* ycur = x_h;  // y(0) = x
  for (int it = 0; it < 10; ++it) {
    gemm_abT_f16<<<dim3(DD / 64, NN / 16, BB), 32, 0, stream>>>(ycur, WQ_h, Q_h, NN, DD, DD);
    attn_iter<<<BB * (NN / 16), 256, smem, stream>>>(K_h, Q_h, xT_h, y32, y_h);
    ycur = y_h;
  }
}
